// AgentModel_38087769981178
// MI455X (gfx1250) — matrix operand order repaired and checked by reading, not compiled, not run on hardware
//
#include <hip/hip_runtime.h>
#include <hip/hip_bf16.h>
#include <math.h>

typedef __bf16 bf16_t;
typedef __attribute__((ext_vector_type(16))) __bf16 bf16x16;
typedef __attribute__((ext_vector_type(8)))  float  floatx8;
typedef __attribute__((ext_vector_type(4)))  int    i32x4;

#define B_DIM  8192
#define F_DIM  1024
#define U_DIM  512
#define O_DIM  3
#define A_DIM  6
#define H_DIM  4
#define HD_DIM 128

// ---------------------------------------------------------------------------
// async global->LDS copy (gfx1250 GLOBAL_LOAD_ASYNC_TO_LDS_B128, ASYNCcnt)
// ---------------------------------------------------------------------------
#if defined(__AMDGCN__) && __has_builtin(__builtin_amdgcn_global_load_async_to_lds_b128)
#define ASYNC_LDS 1
#endif

__device__ __forceinline__ void cp16_g2s(bf16_t* l, const bf16_t* g) {
#ifdef ASYNC_LDS
  __builtin_amdgcn_global_load_async_to_lds_b128((i32x4*)g, (i32x4*)l, 0, 0);
#else
  *(uint4*)l = *(const uint4*)g;
#endif
}

__device__ __forceinline__ void async_wait_all() {
#ifdef ASYNC_LDS
#if __has_builtin(__builtin_amdgcn_s_wait_asynccnt)
  __builtin_amdgcn_s_wait_asynccnt(0);
#else
  asm volatile("s_wait_asynccnt 0x0" ::: "memory");
#endif
#endif
}

// ---------------------------------------------------------------------------
// f32 -> bf16 conversion (vectorized x4)
// ---------------------------------------------------------------------------
__global__ void f32_to_bf16_kernel(const float* __restrict__ src,
                                   bf16_t* __restrict__ dst, int n4) {
  int i = blockIdx.x * blockDim.x + threadIdx.x;
  if (i < n4) {
    float4 f = ((const float4*)src)[i];
    union { bf16_t h[4]; uint2 u; } o;
    o.h[0] = (bf16_t)f.x; o.h[1] = (bf16_t)f.y;
    o.h[2] = (bf16_t)f.z; o.h[3] = (bf16_t)f.w;
    ((uint2*)dst)[i] = o.u;
  }
}

// ---------------------------------------------------------------------------
// Tiled bf16 WMMA GEMM:  C[M,N] = epi( (A[M,K] @ W[N,K]^T + bias) * alpha )
// Block tile 128x128, BK=32, double-buffered LDS, 8 waves, wave grid 2(M)x4(N),
// each wave computes 64x32 = 4x2 tiles of v_wmma_f32_16x16x32_bf16.
// ---------------------------------------------------------------------------
#define LDSW 40                 // padded row stride (bf16 elems)
#define BUFELEMS (128 * LDSW)   // one LDS buffer (128 rows x BK=32, padded)

__device__ __forceinline__ bf16x16 load_frag_a(const bf16_t* rowptr, int lane) {
  // A 16x32 bf16: lane<16 -> K 0..7 (elems 0..7), K 16..23 (elems 8..15)
  //               lane>=16 -> K 8..15, K 24..31
  const int kb = (lane < 16) ? 0 : 8;
  union { bf16x16 v; uint4 u[2]; } f;
  f.u[0] = *(const uint4*)(rowptr + kb);
  f.u[1] = *(const uint4*)(rowptr + kb + 16);
  return f.v;
}

__device__ __forceinline__ bf16x16 load_frag_b(const bf16_t* rowptr, int lane) {
  // B 32x16 bf16: column = lane&15; lane<16 -> K 0..7, K 16..23; lane>=16 -> K 8..15, K 24..31
  const int kb = (lane < 16) ? 0 : 8;
  union { bf16x16 v; uint4 u[2]; } f;
  f.u[0] = *(const uint4*)(rowptr + kb);
  f.u[1] = *(const uint4*)(rowptr + kb + 16);
  return f.v;
}

template <bool ELU, bool OUTBF16>
__global__ __launch_bounds__(256)
void gemm_bf16_kernel(const bf16_t* __restrict__ A, int lda,
                      const bf16_t* __restrict__ W, int ldw,
                      const float* __restrict__ bias, float alpha,
                      void* __restrict__ Cout, int ldc, int K) {
  __shared__ bf16_t sA[2 * BUFELEMS];
  __shared__ bf16_t sB[2 * BUFELEMS];

  const int bn   = blockIdx.x;        // N block (128)
  const int bm   = blockIdx.y;        // M block (128)
  const int lane = threadIdx.x & 31;
  const int wave = threadIdx.x >> 5;  // 0..7
  const int wm   = wave >> 2;         // 0..1  (64-row strip)
  const int wn   = wave & 3;          // 0..3  (32-col strip)

  const floatx8 vzero = {0.f, 0.f, 0.f, 0.f, 0.f, 0.f, 0.f, 0.f};
  floatx8 acc[4][2];
#pragma unroll
  for (int mi = 0; mi < 4; ++mi)
#pragma unroll
    for (int ni = 0; ni < 2; ++ni) acc[mi][ni] = vzero;

  const bf16_t* Ag = A + (size_t)bm * 128 * lda;
  const bf16_t* Wg = W + (size_t)bn * 128 * ldw;

  // stage one 128x32 tile of A and W into LDS buffer `buf`
  auto stage = [&](int buf, int k0) {
#pragma unroll
    for (int i = 0; i < 2; ++i) {
      const int c   = threadIdx.x + i * 256;  // chunk id, 4 chunks/row
      const int row = c >> 2;
      const int col = (c & 3) * 8;
      cp16_g2s(sA + buf * BUFELEMS + row * LDSW + col,
               Ag + (size_t)row * lda + k0 + col);
      cp16_g2s(sB + buf * BUFELEMS + row * LDSW + col,
               Wg + (size_t)row * ldw + k0 + col);
    }
  };

  const int ktiles = K >> 5;
  stage(0, 0);
  async_wait_all();
  __syncthreads();

  for (int t = 0; t < ktiles; ++t) {
    const int cur = t & 1;
    if (t + 1 < ktiles) stage(cur ^ 1, (t + 1) << 5);  // prefetch next tile

    const bf16_t* bufA = sA + cur * BUFELEMS;
    const bf16_t* bufB = sB + cur * BUFELEMS;
    const int r = lane & 15;
    bf16x16 bfrag[2];
#pragma unroll
    for (int ni = 0; ni < 2; ++ni)
      bfrag[ni] = load_frag_b(bufB + (wn * 32 + ni * 16 + r) * LDSW, lane);
#pragma unroll
    for (int mi = 0; mi < 4; ++mi) {
      bf16x16 afrag = load_frag_a(bufA + (wm * 64 + mi * 16 + r) * LDSW, lane);
#pragma unroll
      for (int ni = 0; ni < 2; ++ni)
        acc[mi][ni] = __builtin_amdgcn_wmma_f32_16x16x32_bf16(
            false, afrag, false, bfrag[ni], (short)0, acc[mi][ni], false, false);
    }
    async_wait_all();
    __syncthreads();
  }

  // epilogue: C/D layout -> elem e: M = e + 8*(lane>=16), N = lane&15
  const int nloc = lane & 15;
  const int mhi  = (lane < 16) ? 0 : 8;
#pragma unroll
  for (int mi = 0; mi < 4; ++mi) {
#pragma unroll
    for (int ni = 0; ni < 2; ++ni) {
      const int ncol  = bn * 128 + wn * 32 + ni * 16 + nloc;
      const float bia = bias[ncol];
      const int mrow0 = bm * 128 + wm * 64 + mi * 16 + mhi;
#pragma unroll
      for (int e = 0; e < 8; ++e) {
        float t = (acc[mi][ni][e] + bia) * alpha;
        if (ELU) t = (t > 0.f) ? t : (__expf(t) - 1.f);
        const size_t idx = (size_t)(mrow0 + e) * ldc + ncol;
        if constexpr (OUTBF16) ((bf16_t*)Cout)[idx] = (bf16_t)t;
        else                   ((float*)Cout)[idx]  = t;
      }
    }
  }
}

// ---------------------------------------------------------------------------
// attention: one block (128 thr = 4 waves) per batch row; wave h = head h
// ---------------------------------------------------------------------------
__device__ __forceinline__ float wave_sum(float x) {
#pragma unroll
  for (int m = 16; m >= 1; m >>= 1) x += __shfl_xor(x, m, 32);
  return x;
}

__global__ __launch_bounds__(128)
void attention_kernel(const float* __restrict__ q, const bf16_t* __restrict__ k,
                      const bf16_t* __restrict__ v, bf16_t* __restrict__ attn,
                      float* __restrict__ infl) {
  __shared__ float sw[H_DIM][O_DIM];
  const int b    = blockIdx.x;
  const int h    = threadIdx.x >> 5;
  const int lane = threadIdx.x & 31;
  const int base = b * U_DIM + h * HD_DIM;

  float qv[4];
#pragma unroll
  for (int j = 0; j < 4; ++j) qv[j] = q[base + j * 32 + lane];

  float s[O_DIM];
#pragma unroll
  for (int o = 0; o < O_DIM; ++o) {
    const bf16_t* kr = k + (size_t)o * B_DIM * U_DIM + base;
    float p = 0.f;
#pragma unroll
    for (int j = 0; j < 4; ++j) p += qv[j] * (float)kr[j * 32 + lane];
    s[o] = wave_sum(p);
  }
  const float mx = fmaxf(s[0], fmaxf(s[1], s[2]));
  const float e0 = __expf(s[0] - mx), e1 = __expf(s[1] - mx), e2 = __expf(s[2] - mx);
  const float inv = 1.f / (e0 + e1 + e2);
  const float w0 = e0 * inv, w1 = e1 * inv, w2 = e2 * inv;

  const bf16_t* v0 = v + base;
  const bf16_t* v1 = v + (size_t)1 * B_DIM * U_DIM + base;
  const bf16_t* v2 = v + (size_t)2 * B_DIM * U_DIM + base;
#pragma unroll
  for (int j = 0; j < 4; ++j) {
    const int d = j * 32 + lane;
    float a = w0 * (float)v0[d] + w1 * (float)v1[d] + w2 * (float)v2[d];
    attn[base + d] = (bf16_t)a;
  }
  if (lane == 0) { sw[h][0] = w0; sw[h][1] = w1; sw[h][2] = w2; }
  __syncthreads();
  if (threadIdx.x < O_DIM) {
    const int o = threadIdx.x;
    infl[b * O_DIM + o] = 0.25f * (sw[0][o] + sw[1][o] + sw[2][o] + sw[3][o]);
  }
}

// ---------------------------------------------------------------------------
// skinny heads: one wave per output row, f32 weights
// ---------------------------------------------------------------------------
__global__ __launch_bounds__(256)
void agent_head_kernel(const bf16_t* __restrict__ head,
                       const float* __restrict__ W_ap, const float* __restrict__ b_ap,
                       const float* __restrict__ W_av, const float* __restrict__ b_av,
                       float* __restrict__ pol, float* __restrict__ val) {
  const int wid  = (blockIdx.x * blockDim.x + threadIdx.x) >> 5;  // batch row
  const int lane = threadIdx.x & 31;
  if (wid >= B_DIM) return;
  const bf16_t* hr = head + (size_t)wid * U_DIM;
  float hv[16];
#pragma unroll
  for (int e = 0; e < 16; ++e) hv[e] = (float)hr[e * 32 + lane];

  for (int a = 0; a < A_DIM; ++a) {
    const float* w = W_ap + (size_t)a * U_DIM;
    float p = 0.f;
#pragma unroll
    for (int e = 0; e < 16; ++e) p += hv[e] * w[e * 32 + lane];
    p = wave_sum(p);
    if (lane == 0) pol[wid * A_DIM + a] = p + b_ap[a];
  }
  float p = 0.f;
#pragma unroll
  for (int e = 0; e < 16; ++e) p += hv[e] * W_av[e * 32 + lane];
  p = wave_sum(p);
  if (lane == 0) val[wid] = p + b_av[0];
}

__global__ __launch_bounds__(256)
void opp_head_kernel(const bf16_t* __restrict__ heads,
                     const float* __restrict__ W_op, const float* __restrict__ b_op,
                     const float* __restrict__ W_ov, const float* __restrict__ b_ov,
                     float* __restrict__ pol, float* __restrict__ val) {
  const int gw   = (blockIdx.x * blockDim.x + threadIdx.x) >> 5;  // 0..B*O-1
  const int lane = threadIdx.x & 31;
  if (gw >= B_DIM * O_DIM) return;
  const int o = gw / B_DIM;
  const int b = gw % B_DIM;
  const bf16_t* hr = heads + ((size_t)o * B_DIM + b) * U_DIM;
  float hv[16];
#pragma unroll
  for (int e = 0; e < 16; ++e) hv[e] = (float)hr[e * 32 + lane];

  for (int a = 0; a < A_DIM; ++a) {
    const float* w = W_op + ((size_t)o * A_DIM + a) * U_DIM;
    float p = 0.f;
#pragma unroll
    for (int e = 0; e < 16; ++e) p += hv[e] * w[e * 32 + lane];
    p = wave_sum(p);
    if (lane == 0) pol[(size_t)b * (O_DIM * A_DIM) + o * A_DIM + a] = p + b_op[o * A_DIM + a];
  }
  float p = 0.f;
#pragma unroll
  for (int e = 0; e < 16; ++e) p += hv[e] * W_ov[(size_t)o * U_DIM + e * 32 + lane];
  p = wave_sum(p);
  if (lane == 0) val[b * O_DIM + o] = p + b_ov[o];
}

// ---------------------------------------------------------------------------
// launcher
// ---------------------------------------------------------------------------
extern "C" void kernel_launch(void* const* d_in, const int* in_sizes, int n_in,
                              void* d_out, int out_size, void* d_ws, size_t ws_size,
                              hipStream_t stream) {
  const float* features = (const float*)d_in[0];
  const float* W_al  = (const float*)d_in[1];  const float* b_al = (const float*)d_in[2];
  const float* W_in  = (const float*)d_in[3];  const float* b_in = (const float*)d_in[4];
  const float* W_out = (const float*)d_in[5];  const float* b_out = (const float*)d_in[6];
  const float* W_ah  = (const float*)d_in[7];  const float* b_ah = (const float*)d_in[8];
  const float* W_ap  = (const float*)d_in[9];  const float* b_ap = (const float*)d_in[10];
  const float* W_av  = (const float*)d_in[11]; const float* b_av = (const float*)d_in[12];
  const float* W_ol  = (const float*)d_in[13]; const float* b_ol = (const float*)d_in[14];
  const float* W_oh  = (const float*)d_in[15]; const float* b_oh = (const float*)d_in[16];
  const float* W_op  = (const float*)d_in[17]; const float* b_op = (const float*)d_in[18];
  const float* W_ov  = (const float*)d_in[19]; const float* b_ov = (const float*)d_in[20];

  float* out = (float*)d_out;
  float* agent_policy = out;                       // [B,6]
  float* agent_value  = out + 49152;               // [B,1]
  float* opp_policies = out + 57344;               // [B,3,6]
  float* opp_values   = out + 204800;              // [B,3,1]
  float* influences   = out + 229376;              // [B,3]

  // workspace bump allocator (256B aligned)
  char* wsp = (char*)d_ws;
  size_t off = 0;
  auto alloc = [&](size_t bytes) -> void* {
    void* p = wsp + off;
    off = (off + bytes + 255) & ~(size_t)255;
    return p;
  };
  bf16_t* feat_bf   = (bf16_t*)alloc((size_t)B_DIM * F_DIM * 2);
  bf16_t* Wal_bf    = (bf16_t*)alloc((size_t)U_DIM * F_DIM * 2);
  bf16_t* Win_bf    = (bf16_t*)alloc((size_t)3 * U_DIM * U_DIM * 2);
  bf16_t* Wout_bf   = (bf16_t*)alloc((size_t)U_DIM * U_DIM * 2);
  bf16_t* Wah_bf    = (bf16_t*)alloc((size_t)U_DIM * 2 * U_DIM * 2);
  bf16_t* Wol_bf    = (bf16_t*)alloc((size_t)O_DIM * U_DIM * F_DIM * 2);
  bf16_t* Woh_bf    = (bf16_t*)alloc((size_t)O_DIM * U_DIM * U_DIM * 2);
  bf16_t* cat_bf    = (bf16_t*)alloc((size_t)B_DIM * 2 * U_DIM * 2);  // [latent|attn_out]
  bf16_t* opplat_bf = (bf16_t*)alloc((size_t)O_DIM * B_DIM * U_DIM * 2);
  float*  q_f32     = (float*) alloc((size_t)B_DIM * U_DIM * 4);
  bf16_t* k_bf      = (bf16_t*)alloc((size_t)O_DIM * B_DIM * U_DIM * 2);
  bf16_t* v_bf      = (bf16_t*)alloc((size_t)O_DIM * B_DIM * U_DIM * 2);
  bf16_t* attn_bf   = (bf16_t*)alloc((size_t)B_DIM * U_DIM * 2);
  bf16_t* opph_bf   = (bf16_t*)alloc((size_t)O_DIM * B_DIM * U_DIM * 2);
  bf16_t* ahead_bf  = (bf16_t*)alloc((size_t)B_DIM * U_DIM * 2);
  (void)ws_size; (void)in_sizes; (void)n_in; (void)out_size;

  auto cvt = [&](const float* s, bf16_t* d, int n) {
    const int n4 = n / 4;
    f32_to_bf16_kernel<<<(n4 + 255) / 256, 256, 0, stream>>>(s, d, n4);
  };
  cvt(features, feat_bf, B_DIM * F_DIM);
  cvt(W_al,  Wal_bf,  U_DIM * F_DIM);
  cvt(W_in,  Win_bf,  3 * U_DIM * U_DIM);
  cvt(W_out, Wout_bf, U_DIM * U_DIM);
  cvt(W_ah,  Wah_bf,  U_DIM * 2 * U_DIM);
  cvt(W_ol,  Wol_bf,  O_DIM * U_DIM * F_DIM);
  cvt(W_oh,  Woh_bf,  O_DIM * U_DIM * U_DIM);

  const dim3 blk(256);
  const dim3 g_wide(U_DIM / 128, B_DIM / 128);  // N=512 GEMMs: (4,64)
  const float qscale = 0.08838834764831845f;    // 1/sqrt(128)

  // agent_latent = elu(features @ W_al^T + b_al) -> cat[:, :512] (bf16)
  gemm_bf16_kernel<true, true><<<g_wide, blk, 0, stream>>>(
      feat_bf, F_DIM, Wal_bf, F_DIM, b_al, 1.f, cat_bf, 2 * U_DIM, F_DIM);
  // opp_lat[o] = elu(features @ W_ol[o]^T + b_ol[o])
  for (int o = 0; o < O_DIM; ++o)
    gemm_bf16_kernel<true, true><<<g_wide, blk, 0, stream>>>(
        feat_bf, F_DIM, Wol_bf + (size_t)o * U_DIM * F_DIM, F_DIM,
        b_ol + o * U_DIM, 1.f, opplat_bf + (size_t)o * B_DIM * U_DIM,
        U_DIM, F_DIM);
  // q = (latent @ Wq^T + bq) * 1/sqrt(HD)  (f32 for attention)
  gemm_bf16_kernel<false, false><<<g_wide, blk, 0, stream>>>(
      cat_bf, 2 * U_DIM, Win_bf, U_DIM, b_in, qscale, q_f32, U_DIM, U_DIM);
  // k[o], v[o], opp_heads[o]
  for (int o = 0; o < O_DIM; ++o) {
    const bf16_t* Ao = opplat_bf + (size_t)o * B_DIM * U_DIM;
    gemm_bf16_kernel<false, true><<<g_wide, blk, 0, stream>>>(
        Ao, U_DIM, Win_bf + (size_t)U_DIM * U_DIM, U_DIM, b_in + U_DIM, 1.f,
        k_bf + (size_t)o * B_DIM * U_DIM, U_DIM, U_DIM);
    gemm_bf16_kernel<false, true><<<g_wide, blk, 0, stream>>>(
        Ao, U_DIM, Win_bf + (size_t)2 * U_DIM * U_DIM, U_DIM, b_in + 2 * U_DIM, 1.f,
        v_bf + (size_t)o * B_DIM * U_DIM, U_DIM, U_DIM);
    gemm_bf16_kernel<true, true><<<g_wide, blk, 0, stream>>>(
        Ao, U_DIM, Woh_bf + (size_t)o * U_DIM * U_DIM, U_DIM, b_oh + o * U_DIM, 1.f,
        opph_bf + (size_t)o * B_DIM * U_DIM, U_DIM, U_DIM);
  }
  // attention + softmax + influences
  attention_kernel<<<B_DIM, 128, 0, stream>>>(q_f32, k_bf, v_bf, attn_bf, influences);
  // attn_out = attn @ W_out^T + b_out -> cat[:, 512:]
  gemm_bf16_kernel<false, true><<<g_wide, blk, 0, stream>>>(
      attn_bf, U_DIM, Wout_bf, U_DIM, b_out, 1.f, cat_bf + U_DIM,
      2 * U_DIM, U_DIM);
  // agent_head = elu(cat @ W_ah^T + b_ah)
  gemm_bf16_kernel<true, true><<<g_wide, blk, 0, stream>>>(
      cat_bf, 2 * U_DIM, Wah_bf, 2 * U_DIM, b_ah, 1.f, ahead_bf, U_DIM,
      2 * U_DIM);
  // skinny heads
  agent_head_kernel<<<B_DIM / 8, 256, 0, stream>>>(ahead_bf, W_ap, b_ap, W_av, b_av,
                                                   agent_policy, agent_value);
  opp_head_kernel<<<(B_DIM * O_DIM) / 8, 256, 0, stream>>>(
      opph_bf, W_op, b_op, W_ov, b_ov, opp_policies, opp_values);
}
